// DenseGraphAttentionHead_62474594288303
// MI455X (gfx1250) — matrix operand order repaired and checked by reading, not compiled, not run on hardware
//
#include <hip/hip_runtime.h>
#include <hip/hip_bf16.h>
#include <cstdint>

#define N_NODES 8192
#define IN_DIM  512
#define OUT_DIM 256
#define ALPHA   0.2f

typedef __attribute__((ext_vector_type(16))) __bf16        v16bf;
typedef __attribute__((ext_vector_type(8)))  float         v8f;
typedef __attribute__((ext_vector_type(4)))  unsigned int  v4u;
typedef __attribute__((ext_vector_type(8)))  int           v8i;
typedef __attribute__((ext_vector_type(4)))  int           v4i;

struct Pair32 { uint4 lo, hi; };   // 32 bytes -> bit_cast to v16bf

static __device__ __forceinline__ uint16_t f32_to_bf16(float f) {
    union { float f; uint32_t u; } v; v.f = f;
    uint32_t u = v.u;
    uint32_t r = u + 0x7FFFu + ((u >> 16) & 1u);   // round-to-nearest-even
    return (uint16_t)(r >> 16);
}
static __device__ __forceinline__ uint32_t pack_bf16(float a, float b) {
    return (uint32_t)f32_to_bf16(a) | ((uint32_t)f32_to_bf16(b) << 16);
}
static __device__ __forceinline__ v16bf make_v16bf(uint4 lo, uint4 hi) {
    Pair32 p; p.lo = lo; p.hi = hi;
    return __builtin_bit_cast(v16bf, p);
}
static __device__ __forceinline__ float lrelu(float x) {
    return x > 0.f ? x : ALPHA * x;
}

// TDM: async 2D tile load  WhT[0..255][jb..jb+32) -> LDS (256 rows x 64 B).
static __device__ __forceinline__ void tdm_load_b_tile(
    const uint16_t* WhT, int jb, uint32_t lds_addr)
{
    const uint64_t ga = (uint64_t)(uintptr_t)WhT + (uint64_t)jb * 2u;
    v4u g0;
    g0[0] = 1u;                                        // count=1 (valid D#)
    g0[1] = lds_addr;                                  // lds_addr
    g0[2] = (uint32_t)ga;                              // global_addr[31:0]
    g0[3] = (uint32_t)((ga >> 32) & 0x01FFFFFFu) | (2u << 30); // addr[56:32] | type=2
    v8i g1;
    g1[0] = (int)(1u << 16);                           // data_size = 2 bytes
    g1[1] = (int)((uint32_t)(N_NODES & 0xFFFF) << 16); // tensor_dim0[15:0]
    g1[2] = (int)(((uint32_t)N_NODES >> 16) |
                  ((uint32_t)(OUT_DIM & 0xFFFF) << 16)); // dim0[31:16] | dim1[15:0]
    g1[3] = (int)(32u << 16);                          // dim1[31:16]=0 | tile_dim0=32
    g1[4] = (int)OUT_DIM;                              // tile_dim1=256, tile_dim2=0
    g1[5] = (int)N_NODES;                              // tensor_dim0_stride[31:0]
    g1[6] = 0;
    g1[7] = 0;
    v4i z4 = {};
    v8i z8 = {};
    __builtin_amdgcn_tensor_load_to_lds(g0, g1, z4, z4, z8, 0);
}

// ---------------------------------------------------------------------------
// Kernel 1: Wh = nodes @ W_w.T + W_b  (bf16 WMMA, fp32 accum)
// Writes WhT (column-major bf16 [OUT_DIM][N]) and raw attention scalars
// s1[j] = Wh[j].a1 + a1b,  s2[j] = Wh[j].a2 + a2b  via LDS reduction.
// Block: 256 thr = 8 waves; tile 64 rows x 256 cols; K step 32, double-buffered.
// ---------------------------------------------------------------------------
__global__ __launch_bounds__(256) void wh_kernel(
    const float* __restrict__ nodes, const float* __restrict__ Ww,
    const float* __restrict__ Wb, const float* __restrict__ a1w,
    const float* __restrict__ a1b, const float* __restrict__ a2w,
    const float* __restrict__ a2b,
    uint16_t* __restrict__ WhT, float* __restrict__ s1, float* __restrict__ s2)
{
    __shared__ uint32_t ldsA[2][64 * 16];    // A tile: 64 x 32 bf16 (row-major)
    __shared__ uint32_t ldsB[2][256 * 16];   // B tile: [n][k] 256 x 32 bf16
    __shared__ float s1p[64], s2p[64];

    const int t  = threadIdx.x;
    const int w  = t >> 5, l = t & 31;
    const int rg = w & 3, ch = w >> 2;    // row-group (16 rows), col-half (128 cols)
    const int lm = l & 15, kh = l >> 4;
    const int Mbase = blockIdx.x * 64;

    if (t < 64) { s1p[t] = 0.f; s2p[t] = 0.f; }

    v8f acc[8] = {};
    int ph = 0;

    for (int kb = 0; kb < IN_DIM; kb += 32, ph ^= 1) {
        // --- stage A: nodes[Mbase..+64][kb..+32] -> bf16 LDS (8 floats/thread)
        {
            const int row = t >> 2, ks = (t & 3) * 8;
            const float* src = nodes + (size_t)(Mbase + row) * IN_DIM + kb + ks;
            float4 f0 = *(const float4*)(src);
            float4 f1 = *(const float4*)(src + 4);
            uint4 o; o.x = pack_bf16(f0.x, f0.y); o.y = pack_bf16(f0.z, f0.w);
            o.z = pack_bf16(f1.x, f1.y); o.w = pack_bf16(f1.z, f1.w);
            *(uint4*)&ldsA[ph][row * 16 + (ks >> 1)] = o;
        }
        // --- stage B: ldsB[n][k] = Ww[n][kb+k]  (thread t handles n = t)
        {
            const float* src = Ww + (size_t)t * IN_DIM + kb;
            #pragma unroll
            for (int q = 0; q < 4; ++q) {
                float4 f0 = *(const float4*)(src + q * 8);
                float4 f1 = *(const float4*)(src + q * 8 + 4);
                uint4 o; o.x = pack_bf16(f0.x, f0.y); o.y = pack_bf16(f0.z, f0.w);
                o.z = pack_bf16(f1.x, f1.y); o.w = pack_bf16(f1.z, f1.w);
                *(uint4*)&ldsB[ph][t * 16 + q * 4] = o;
            }
        }
        __syncthreads();
        // --- consume: A once, B operands ping-ponged ahead of each WMMA
        const uint32_t* pA = &ldsA[ph][0];
        const uint32_t* pB = &ldsB[ph][0];
        const int arow = 16 * rg + lm;
        uint4 alo = *(const uint4*)&pA[arow * 16 + kh * 8];
        uint4 ahi = *(const uint4*)&pA[arow * 16 + kh * 8 + 4];
        v16bf a = make_v16bf(alo, ahi);
        const int nb = (ch * 128 + lm) * 16 + kh * 8;   // dword index, s=0
        uint4 blo = *(const uint4*)&pB[nb];
        uint4 bhi = *(const uint4*)&pB[nb + 4];
        v16bf bc = make_v16bf(blo, bhi);
        #pragma unroll
        for (int s = 0; s < 8; ++s) {
            v16bf bn = bc;
            if (s < 7) {
                uint4 nlo = *(const uint4*)&pB[nb + (s + 1) * 256];
                uint4 nhi = *(const uint4*)&pB[nb + (s + 1) * 256 + 4];
                bn = make_v16bf(nlo, nhi);
            }
            acc[s] = __builtin_amdgcn_wmma_f32_16x16x32_bf16(
                false, a, false, bc, (short)0, acc[s], false, false);
            bc = bn;
        }
    }

    // --- epilogue: bias, column-major bf16 store, s1/s2 partial reductions
    #pragma unroll
    for (int s = 0; s < 8; ++s) {
        const int n = ch * 128 + s * 16 + lm;
        const float bias = Wb[n];
        const float w1 = a1w[n], w2 = a2w[n];
        float vals[8];
        #pragma unroll
        for (int v = 0; v < 8; ++v) vals[v] = acc[s][v] + bias;
        uint4 st; st.x = pack_bf16(vals[0], vals[1]); st.y = pack_bf16(vals[2], vals[3]);
        st.z = pack_bf16(vals[4], vals[5]); st.w = pack_bf16(vals[6], vals[7]);
        const int j0 = Mbase + 16 * rg + 8 * kh;    // acc VGPR v -> consecutive j
        *(uint4*)&WhT[(size_t)n * N_NODES + j0] = st;
        #pragma unroll
        for (int v = 0; v < 8; ++v) {
            const int rloc = 16 * rg + v + 8 * kh;
            atomicAdd(&s1p[rloc], w1 * vals[v]);
            atomicAdd(&s2p[rloc], w2 * vals[v]);
        }
    }
    __syncthreads();
    if (t < 64) {
        s1[Mbase + t] = s1p[t] + a1b[0];
        s2[Mbase + t] = s2p[t] + a2b[0];
    }
}

// ---------------------------------------------------------------------------
// Kernel 2: per-row masked softmax stats (max, 1/sum). One wave per row.
// ---------------------------------------------------------------------------
__global__ __launch_bounds__(256) void rowstats_kernel(
    const unsigned char* __restrict__ edge, const float* __restrict__ s1,
    const float* __restrict__ s2, float* __restrict__ rowm,
    float* __restrict__ rowinv)
{
    const int w = threadIdx.x >> 5, l = threadIdx.x & 31;
    const int i = blockIdx.x * 8 + w;
    const float s1i = s1[i];
    const unsigned char* erow = edge + (size_t)i * N_NODES;

    float m = -INFINITY, s = 0.f;
    for (int c = 0; c < N_NODES / 128; ++c) {
        const int j0 = c * 128 + l * 4;
        const uint32_t e4 = *(const uint32_t*)(erow + j0);
        const float4 sv = *(const float4*)(s2 + j0);
        const float xs[4] = { sv.x, sv.y, sv.z, sv.w };
        #pragma unroll
        for (int q = 0; q < 4; ++q) {
            if ((e4 >> (8 * q)) & 0xffu) {
                const float x = lrelu(s1i + xs[q]);
                const float mn = fmaxf(m, x);
                s = s * __expf(m - mn) + __expf(x - mn);
                m = mn;
            }
        }
    }
    #pragma unroll
    for (int off = 16; off >= 1; off >>= 1) {
        const float mo = __shfl_xor(m, off, 32);
        const float so = __shfl_xor(s, off, 32);
        const float mn = fmaxf(m, mo);
        s = (mn == -INFINITY) ? 0.f
                              : s * __expf(m - mn) + so * __expf(mo - mn);
        m = mn;
    }
    if (l == 0) { rowm[i] = m; rowinv[i] = 1.f / s; }
}

// ---------------------------------------------------------------------------
// Kernel 3: out = softmax(scores) @ Wh, fused. Block tile 64 rows x 256 cols.
// B tile staged by the Tensor Data Mover, pipelined one K-step ahead so the
// 16 KB DMA overlaps the WMMA chain; P tile built on the fly. Double-buffered
// LDS, one barrier per K-step.
// ---------------------------------------------------------------------------
__global__ __launch_bounds__(256) void attn_kernel(
    const unsigned char* __restrict__ edge, const uint16_t* __restrict__ WhT,
    const float* __restrict__ s1, const float* __restrict__ s2,
    const float* __restrict__ rowm, const float* __restrict__ rowinv,
    float* __restrict__ out)
{
    __shared__ uint32_t ldsP[2][64 * 16];    // P tile: 64 x 32 bf16
    __shared__ uint32_t ldsB[2][256 * 16];   // B tile: [n][k] 256 x 32 bf16
    __shared__ float sh_s1[64], sh_m[64], sh_inv[64];

    const int t  = threadIdx.x;
    const int w  = t >> 5, l = t & 31;
    const int rg = w & 3, ch = w >> 2;
    const int lm = l & 15, kh = l >> 4;
    const int Mbase = blockIdx.x * 64;

    if (t < 64) {
        sh_s1[t]  = s1[Mbase + t];
        sh_m[t]   = rowm[Mbase + t];
        sh_inv[t] = rowinv[Mbase + t];
    }
    __syncthreads();

    v8f acc[8] = {};

    const int prow = t >> 2;          // local row for P staging
    const int pk   = (t & 3) * 8;     // 8 consecutive j per thread
    const unsigned char* erowp = edge + (size_t)(Mbase + prow) * N_NODES + pk;

    // LDS byte addresses of the two B buffers (flat addr low 32 bits = LDS offset)
    const uint32_t ldsB_addr0 = (uint32_t)(uintptr_t)&ldsB[0][0];
    const uint32_t ldsB_addr1 = (uint32_t)(uintptr_t)&ldsB[1][0];

    // Prologue: kick off the first B-tile DMA (-> buffer 0).
    if (w == 0) tdm_load_b_tile(WhT, 0, ldsB_addr0);

    int ph = 0;
    for (int jb = 0; jb < N_NODES; jb += 32, ph ^= 1) {
        // --- build P tile: p = edge ? exp(lrelu(s1+s2) - m) * inv : 0 --------
        {
            const unsigned long long e8 = *(const unsigned long long*)(erowp + jb);
            if (jb + 32 < N_NODES)
                __builtin_prefetch((const void*)(erowp + jb + 32), 0, 1);
            const float4 svA = *(const float4*)(s2 + jb + pk);
            const float4 svB = *(const float4*)(s2 + jb + pk + 4);
            const float m_i = sh_m[prow], inv = sh_inv[prow], s1v = sh_s1[prow];
            const float xs[8] = { svA.x, svA.y, svA.z, svA.w,
                                  svB.x, svB.y, svB.z, svB.w };
            float p[8];
            #pragma unroll
            for (int q = 0; q < 8; ++q) {
                const bool e = ((e8 >> (8 * q)) & 0xffull) != 0;
                const float x = lrelu(s1v + xs[q]);
                p[q] = e ? __expf(x - m_i) * inv : 0.f;
            }
            uint4 o; o.x = pack_bf16(p[0], p[1]); o.y = pack_bf16(p[2], p[3]);
            o.z = pack_bf16(p[4], p[5]); o.w = pack_bf16(p[6], p[7]);
            *(uint4*)&ldsP[ph][prow * 16 + (pk >> 1)] = o;
        }
        // Current B tile must be fully in LDS before the barrier.
        if (w == 0) __builtin_amdgcn_s_wait_tensorcnt(0);
        __syncthreads();
        // Kick off next tile's DMA into the other buffer; overlaps the WMMAs.
        // (Safe: that buffer's readers finished before the barrier above.)
        if (w == 0 && jb + 32 < N_NODES)
            tdm_load_b_tile(WhT, jb + 32, ph ? ldsB_addr0 : ldsB_addr1);
        // --- WMMA chain: B operands ping-ponged one sub-tile ahead -----------
        const uint32_t* pP = &ldsP[ph][0];
        const uint32_t* pB = &ldsB[ph][0];
        const int arow = 16 * rg + lm;
        uint4 alo = *(const uint4*)&pP[arow * 16 + kh * 8];
        uint4 ahi = *(const uint4*)&pP[arow * 16 + kh * 8 + 4];
        v16bf a = make_v16bf(alo, ahi);
        const int nb = (ch * 128 + lm) * 16 + kh * 8;   // dword index, s=0
        uint4 blo = *(const uint4*)&pB[nb];
        uint4 bhi = *(const uint4*)&pB[nb + 4];
        v16bf bc = make_v16bf(blo, bhi);
        #pragma unroll
        for (int s = 0; s < 8; ++s) {
            v16bf bn = bc;
            if (s < 7) {
                uint4 nlo = *(const uint4*)&pB[nb + (s + 1) * 256];
                uint4 nhi = *(const uint4*)&pB[nb + (s + 1) * 256 + 4];
                bn = make_v16bf(nlo, nhi);
            }
            acc[s] = __builtin_amdgcn_wmma_f32_16x16x32_bf16(
                false, a, false, bc, (short)0, acc[s], false, false);
            bc = bn;
        }
    }

    // --- write output (fp32 row-major [N][256])
    #pragma unroll
    for (int s = 0; s < 8; ++s) {
        const int n = ch * 128 + s * 16 + lm;
        #pragma unroll
        for (int v = 0; v < 8; ++v) {
            const int row = Mbase + 16 * rg + v + 8 * kh;
            out[(size_t)row * OUT_DIM + n] = acc[s][v];
        }
    }
}

// ---------------------------------------------------------------------------
extern "C" void kernel_launch(void* const* d_in, const int* in_sizes, int n_in,
                              void* d_out, int out_size, void* d_ws, size_t ws_size,
                              hipStream_t stream) {
    const float*         nodes = (const float*)d_in[0];
    const unsigned char* edge  = (const unsigned char*)d_in[1];  // bool [N,N]
    const float*         Ww    = (const float*)d_in[2];
    const float*         Wb    = (const float*)d_in[3];
    const float*         a1w   = (const float*)d_in[4];
    const float*         a1b   = (const float*)d_in[5];
    const float*         a2w   = (const float*)d_in[6];
    const float*         a2b   = (const float*)d_in[7];
    float* out = (float*)d_out;

    // workspace carve-up
    uint16_t* WhT  = (uint16_t*)d_ws;                                  // 4 MB
    float* s1      = (float*)((char*)d_ws + (size_t)OUT_DIM * N_NODES * 2);
    float* s2      = s1 + N_NODES;
    float* rowm    = s2 + N_NODES;
    float* rowinv  = rowm + N_NODES;

    wh_kernel<<<N_NODES / 64, 256, 0, stream>>>(nodes, Ww, Wb, a1w, a1b,
                                                a2w, a2b, WhT, s1, s2);
    rowstats_kernel<<<N_NODES / 8, 256, 0, stream>>>(edge, s1, s2, rowm, rowinv);
    attn_kernel<<<N_NODES / 64, 256, 0, stream>>>(edge, WhT, s1, s2,
                                                  rowm, rowinv, out);
}
